// PINN_Euler_12154757448259
// MI455X (gfx1250) — matrix operand order repaired and checked by reading, not compiled, not run on hardware
//
#include <hip/hip_runtime.h>

typedef _Float16 v8h  __attribute__((ext_vector_type(8)));
typedef _Float16 v16h __attribute__((ext_vector_type(16)));
typedef float    v8f  __attribute__((ext_vector_type(8)));

union F16x16 { v16h v; v8h h[2]; };

#define WMMA_F16(a, b, c) \
  __builtin_amdgcn_wmma_f32_16x16x32_f16(false, (a), false, (b), (short)0, (c), false, false)

// ---- LDS layout (bytes). Rows padded to 136 halves (272 B) to spread banks. ----
#define WSTR        136                     // padded row stride in halves
#define OFF_W123    0                       // 3 * 128*WSTR f16  = 104448
#define OFF_W4      104448                  // 16*WSTR f16       = 4352
#define OFF_W0      108800                  // 3*128 f32         = 1536
#define OFF_B       110336                  // 4*128 f32 (b0..b3)= 2048
#define OFF_B4      112384                  // 16 f32            = 64
#define OFF_STAGE   112448                  // 8 waves*4 streams*16*WSTR f16 = 139264
#define OFF_OSTAGE  251712                  // 8 waves*16*16 f32 = 8192
#define LDS_BYTES   259904

__global__ __launch_bounds__(256)
void pinn_euler_wmma(const float* __restrict__ gt, const float* __restrict__ gx,
                     const float* __restrict__ gy,
                     const float* __restrict__ W0, const float* __restrict__ b0,
                     const float* __restrict__ W1, const float* __restrict__ b1,
                     const float* __restrict__ W2, const float* __restrict__ b2,
                     const float* __restrict__ W3, const float* __restrict__ b3,
                     const float* __restrict__ W4, const float* __restrict__ b4,
                     float* __restrict__ out)
{
    extern __shared__ char smem[];
    _Float16* Wl  = (_Float16*)(smem + OFF_W123);   // col-major f16: Wl[L][n*WSTR + k]
    _Float16* W4l = (_Float16*)(smem + OFF_W4);     // [16][WSTR], cols 4..15 zero
    float*    W0l = (float*)(smem + OFF_W0);        // row-major [3][128]
    float*    bl  = (float*)(smem + OFF_B);         // b0,b1,b2,b3
    float*    b4l = (float*)(smem + OFF_B4);        // padded to 16
    _Float16* stg = (_Float16*)(smem + OFF_STAGE);
    float*    ost = (float*)(smem + OFF_OSTAGE);

    const int tid = threadIdx.x;

    // ------------- preload weights: f32 -> f16, transpose to [n][k] -------------
    {
        const float* srcs[3] = {W1, W2, W3};
        for (int L = 0; L < 3; ++L) {
            const float* src = srcs[L];
            _Float16* dst = Wl + L * (128 * WSTR);
            #pragma unroll 8
            for (int j = 0; j < 64; ++j) {
                int i = tid + j * 256;            // 0..16383, coalesced read
                int k = i >> 7, n = i & 127;      // global W[k][n] row-major
                dst[n * WSTR + k] = (_Float16)src[i];
            }
        }
        #pragma unroll
        for (int j = 0; j < 8; ++j) {
            int i = tid + j * 256;                // 0..2047
            int n = i >> 7, k = i & 127;
            W4l[n * WSTR + k] = (n < 4) ? (_Float16)W4[k * 4 + n] : (_Float16)0.0f;
        }
        #pragma unroll
        for (int j = 0; j < 2; ++j) {
            int i = tid + j * 256;
            if (i < 384) W0l[i] = W0[i];
        }
        if (tid < 128) {
            bl[tid]       = b0[tid];
            bl[128 + tid] = b1[tid];
            bl[256 + tid] = b2[tid];
            bl[384 + tid] = b3[tid];
        }
        if (tid < 16) b4l[tid] = (tid < 4) ? b4[tid] : 0.0f;
    }
    __syncthreads();

    const int w   = tid >> 5;
    const int ln  = tid & 31;
    const int lg  = ln >> 4;       // lane half-group
    const int li  = ln & 15;
    const int rb  = lg * 8;        // D-layout row base for this lane group
    const int pbase = blockIdx.x * 128 + w * 16;

    _Float16* stH = stg + (w * 4 + 0) * (16 * WSTR);
    _Float16* stT = stg + (w * 4 + 1) * (16 * WSTR);
    _Float16* stX = stg + (w * 4 + 2) * (16 * WSTR);
    _Float16* stY = stg + (w * 4 + 3) * (16 * WSTR);

    // A-fragment gather from a row-major 16xWSTR staging tile (ISA 16-bit A 16x32 layout)
    auto loadA = [&](const _Float16* st, int kc) -> v16h {
        const _Float16* p = st + li * WSTR + kc * 32 + lg * 8;
        F16x16 r;
        r.h[0] = *(const v8h*)(p);        // K .. K+7   -> VGPR 0..3
        r.h[1] = *(const v8h*)(p + 16);   // K+16..K+23 -> VGPR 4..7
        return r.v;
    };

    // ------------- layer 0 (3 -> 128) directly in D-layout via VALU -------------
    float tr[8], xr[8], yr[8];
    #pragma unroll
    for (int v = 0; v < 8; ++v) {
        int p = pbase + rb + v;
        tr[v] = gt[p]; xr[v] = gx[p]; yr[v] = gy[p];
    }
    #pragma unroll
    for (int ct = 0; ct < 8; ++ct) {
        int n = ct * 16 + li;
        float w0t = W0l[n], w0x = W0l[128 + n], w0y = W0l[256 + n], bb = bl[n];
        #pragma unroll
        for (int v = 0; v < 8; ++v) {
            float z = fmaf(tr[v], w0t, fmaf(xr[v], w0x, fmaf(yr[v], w0y, bb)));
            float c = __cosf(z);
            int idx = (rb + v) * WSTR + n;
            stH[idx] = (_Float16)__sinf(z);
            stT[idx] = (_Float16)(c * w0t);   // dz/dt = W0[0][n]
            stX[idx] = (_Float16)(c * w0x);
            stY[idx] = (_Float16)(c * w0y);
        }
    }

    v16h aH[4], aT[4], aX[4], aY[4];
    #pragma unroll
    for (int kc = 0; kc < 4; ++kc) {
        aH[kc] = loadA(stH, kc); aT[kc] = loadA(stT, kc);
        aX[kc] = loadA(stX, kc); aY[kc] = loadA(stY, kc);
    }

    // tangent helper: dh_next = cos(z) * (dh @ W), stored back to staging as f16
    auto tangent = [&](const v16h (&A)[4], const v16h (&B)[4],
                       const float (&cv)[8], _Float16* S, int ct) {
        v8f a2 = {0.f, 0.f, 0.f, 0.f, 0.f, 0.f, 0.f, 0.f};
        #pragma unroll
        for (int kc = 0; kc < 4; ++kc) a2 = WMMA_F16(A[kc], B[kc], a2);
        #pragma unroll
        for (int v = 0; v < 8; ++v)
            S[(rb + v) * WSTR + ct * 16 + li] = (_Float16)(cv[v] * a2[v]);
    };

    // ------------- hidden layers 1..3 (128 -> 128), WMMA -------------
    for (int L = 0; L < 3; ++L) {
        const _Float16* WB = Wl + L * (128 * WSTR);
        const float* bArr = bl + (L + 1) * 128;
        #pragma unroll
        for (int ct = 0; ct < 8; ++ct) {
            // B fragments: col n = ct*16+li, lanes 0-15 hold K 0..7,16..23, 16-31 hold K 8..15,24..31
            v16h B[4];
            #pragma unroll
            for (int kc = 0; kc < 4; ++kc) {
                const _Float16* cp = WB + (ct * 16 + li) * WSTR + kc * 32 + lg * 8;
                F16x16 r;
                r.h[0] = *(const v8h*)(cp);
                r.h[1] = *(const v8h*)(cp + 16);
                B[kc] = r.v;
            }
            // primal: accumulate from inline 0, fold bias into epilogue so the
            // first WMMA never waits on the bias load / splat.
            v8f acc = {0.f, 0.f, 0.f, 0.f, 0.f, 0.f, 0.f, 0.f};
            #pragma unroll
            for (int kc = 0; kc < 4; ++kc) acc = WMMA_F16(aH[kc], B[kc], acc);
            float bb = bArr[ct * 16 + li];
            float cv[8];
            #pragma unroll
            for (int v = 0; v < 8; ++v) {
                float z = acc[v] + bb;
                stH[(rb + v) * WSTR + ct * 16 + li] = (_Float16)__sinf(z);
                cv[v] = __cosf(z);
            }
            tangent(aT, B, cv, stT, ct);
            tangent(aX, B, cv, stX, ct);
            tangent(aY, B, cv, stY, ct);
        }
        #pragma unroll
        for (int kc = 0; kc < 4; ++kc) {
            aH[kc] = loadA(stH, kc); aT[kc] = loadA(stT, kc);
            aX[kc] = loadA(stX, kc); aY[kc] = loadA(stY, kc);
        }
    }

    // ------------- output layer (128 -> 4, zero-padded to 16 cols) -------------
    {
        v16h B[4];
        #pragma unroll
        for (int kc = 0; kc < 4; ++kc) {
            const _Float16* cp = W4l + li * WSTR + kc * 32 + lg * 8;
            F16x16 r;
            r.h[0] = *(const v8h*)(cp);
            r.h[1] = *(const v8h*)(cp + 16);
            B[kc] = r.v;
        }
        v8f aP  = {0.f,0.f,0.f,0.f,0.f,0.f,0.f,0.f};
        v8f aTt = aP, aXt = aP, aYt = aP;
        #pragma unroll
        for (int kc = 0; kc < 4; ++kc) {
            aP  = WMMA_F16(aH[kc], B[kc], aP);
            aTt = WMMA_F16(aT[kc], B[kc], aTt);
            aXt = WMMA_F16(aX[kc], B[kc], aXt);
            aYt = WMMA_F16(aY[kc], B[kc], aYt);
        }
        float bb = b4l[li];                     // bias only on primal output
        float* ow = ost + w * 256;              // [16 rows][4 streams * 4 comps]
        if (li < 4) {
            #pragma unroll
            for (int v = 0; v < 8; ++v) {
                int r = rb + v;
                ow[r * 16 + 0  + li] = aP[v] + bb;
                ow[r * 16 + 4  + li] = aTt[v];
                ow[r * 16 + 8  + li] = aXt[v];
                ow[r * 16 + 12 + li] = aYt[v];
            }
        }
    }

    // ------------- Euler flux residual per point (lanes 0..15) -------------
    if (ln < 16) {
        const float* os = ost + w * 256 + ln * 16;
        float rho = os[0],  u  = os[1],  v  = os[2],  p  = os[3];
        float rt  = os[4],  ut = os[5],  vt = os[6],  pt = os[7];
        float rx  = os[8],  ux = os[9],  vx = os[10], px = os[11];
        float ry  = os[12], uy = os[13], vy = os[14], py = os[15];
        const float ig = 2.5f;                  // 1/(gamma-1), gamma=1.4
        float q  = u * u + v * v;
        float E  = 0.5f * rho * q + p * ig;
        float Et = 0.5f * rt * q + rho * (u * ut + v * vt) + pt * ig;
        float Ex = 0.5f * rx * q + rho * (u * ux + v * vx) + px * ig;
        float Ey = 0.5f * ry * q + rho * (u * uy + v * vy) + py * ig;
        // U_t
        float r0 = rt;
        float r1 = rt * u + rho * ut;
        float r2 = rt * v + rho * vt;
        float r3 = Et;
        // F_x
        r0 += rx * u + rho * ux;
        r1 += rx * u * u + 2.0f * rho * u * ux + px;
        r2 += rx * u * v + rho * (ux * v + u * vx);
        r3 += (Ex + px) * u + (E + p) * ux;
        // G_y
        r0 += ry * v + rho * vy;
        r1 += ry * u * v + rho * (uy * v + u * vy);
        r2 += ry * v * v + 2.0f * rho * v * vy + py;
        r3 += (Ey + py) * v + (E + p) * vy;

        int pidx = pbase + ln;
        float4 res = make_float4(r0, r1, r2, r3);
        *(float4*)(out + pidx * 4) = res;
    }
}

extern "C" void kernel_launch(void* const* d_in, const int* in_sizes, int n_in,
                              void* d_out, int out_size, void* d_ws, size_t ws_size,
                              hipStream_t stream) {
    const float* t  = (const float*)d_in[0];
    const float* x  = (const float*)d_in[1];
    const float* y  = (const float*)d_in[2];
    const float* W0 = (const float*)d_in[3];
    const float* b0 = (const float*)d_in[4];
    const float* W1 = (const float*)d_in[5];
    const float* b1 = (const float*)d_in[6];
    const float* W2 = (const float*)d_in[7];
    const float* b2 = (const float*)d_in[8];
    const float* W3 = (const float*)d_in[9];
    const float* b3 = (const float*)d_in[10];
    const float* W4 = (const float*)d_in[11];
    const float* b4 = (const float*)d_in[12];
    float* out = (float*)d_out;

    int n = in_sizes[0];          // 262144 points, divisible by 128
    int grid = n >> 7;            // 128 points per block (8 waves x 16 points)
    pinn_euler_wmma<<<grid, 256, LDS_BYTES, stream>>>(
        t, x, y, W0, b0, W1, b1, W2, b2, W3, b3, W4, b4, out);
}
